// PureCorrelation_80796924773031
// MI455X (gfx1250) — matrix operand order repaired and checked by reading, not compiled, not run on hardware
//
#include <hip/hip_runtime.h>

typedef __bf16 bf16_t;
typedef __attribute__((ext_vector_type(16))) __bf16 v16bf;
typedef __attribute__((ext_vector_type(8)))  __bf16 v8bf;
typedef __attribute__((ext_vector_type(8)))  float  v8f;
typedef __attribute__((ext_vector_type(4)))  float  v4f;

#define B_ 4
#define S_ 4096
#define E_ 512

// ---------------- fragment helpers (CDNA5 wave32 WMMA layouts) ----------------

// A-matrix 16x32 bf16: lane holds row m = lane&15; element i -> K = base + i,
// base = 16 for lanes 16..31. Source is f32 row-major; convert inline.
__device__ inline v16bf a_frag_from_f32(const float* __restrict__ row, int kOff, int lane) {
  const int b8 = (lane & 16) ? 16 : 0;
  const v4f q0 = *(const v4f*)(row + kOff + b8);
  const v4f q1 = *(const v4f*)(row + kOff + b8 + 4);
  const v4f q2 = *(const v4f*)(row + kOff + b8 + 8);
  const v4f q3 = *(const v4f*)(row + kOff + b8 + 12);
  v16bf a;
#pragma unroll
  for (int i = 0; i < 4; ++i) {
    a[i]      = (bf16_t)q0[i];
    a[4 + i]  = (bf16_t)q1[i];
    a[8 + i]  = (bf16_t)q2[i];
    a[12 + i] = (bf16_t)q3[i];
  }
  return a;
}

// B-matrix 32x16 bf16: lane holds col n = lane&15; element i -> K = bb + i, bb = 16 for lanes 16..31.
// btRow points at row n of B^T (contiguous in K). bf16 source: one 32B load.
__device__ inline v16bf b_frag_from_bf16(const bf16_t* __restrict__ btRow, int kOff, int lane) {
  const int bb = (lane & 16) ? 16 : 0;
  return *(const v16bf*)(btRow + kOff + bb);
}

// Same but from f32 source with inline conversion.
__device__ inline v16bf b_frag_from_f32(const float* __restrict__ btRow, int kOff, int lane) {
  const int bb = (lane & 16) ? 16 : 0;
  const float* p = btRow + kOff + bb;
  const v4f q0 = *(const v4f*)(p);
  const v4f q1 = *(const v4f*)(p + 4);
  const v4f q2 = *(const v4f*)(p + 8);
  const v4f q3 = *(const v4f*)(p + 12);
  v16bf b;
#pragma unroll
  for (int i = 0; i < 4; ++i) {
    b[i]      = (bf16_t)q0[i];
    b[4 + i]  = (bf16_t)q1[i];
    b[8 + i]  = (bf16_t)q2[i];
    b[12 + i] = (bf16_t)q3[i];
  }
  return b;
}

__device__ inline v8f wmma_bf16(v16bf a, v16bf b, v8f c) {
  return __builtin_amdgcn_wmma_f32_16x16x32_bf16(false, a, false, b, (short)0, c, false, false);
}

// ---------------- phase 1a: k_bf[b][s][f] = bf16( key @ W1^T + b1 ) ----------------
// Tile computed as k^T = W1 @ key^T  (A rows = W1 rows, Bt rows = key rows),
// C layout (lane = s-col, vgpr r -> f-row) => contiguous 8-bf16 row-major store of k.
__global__ __launch_bounds__(128) void proj_k_kernel(const float* __restrict__ key,
                                                     const float* __restrict__ W1,
                                                     const float* __restrict__ b1,
                                                     bf16_t* __restrict__ kbf) {
  const int lane = threadIdx.x & 31;
  const int wave = threadIdx.x >> 5;
  const int tilesF = E_ / 16;               // 32
  const int tilesS = S_ / 16;               // 256
  const int perBatch = tilesF * tilesS;
  int tile = blockIdx.x * 4 + wave;
  const int b  = tile / perBatch;
  const int rem = tile % perBatch;
  const int fT = rem % tilesF;
  const int sT = rem / tilesF;

  const float* aRow  = W1  + (size_t)(fT * 16 + (lane & 15)) * E_;
  const float* btRow = key + ((size_t)b * S_ + sT * 16 + (lane & 15)) * E_;

  v8f c = {};
#pragma unroll
  for (int kk = 0; kk < E_; kk += 32) {
    v16bf a  = a_frag_from_f32(aRow, kk, lane);
    v16bf bb = b_frag_from_f32(btRow, kk, lane);
    c = wmma_bf16(a, bb, c);
  }

  const int fLo = fT * 16 + ((lane & 16) ? 8 : 0);
  bf16_t* outP = kbf + ((size_t)b * S_ + sT * 16 + (lane & 15)) * E_ + fLo;
  v8bf o;
#pragma unroll
  for (int r = 0; r < 8; ++r) o[r] = (bf16_t)(c[r] + b1[fLo + r]);
  *(v8bf*)outP = o;
}

// ---------------- phase 1b: vT[b][e][s] = bf16( (value @ W2^T + b2)^T ) ----------------
// A rows = value rows (m = s), Bt rows = W2 rows (n = e). C store is contiguous in s.
__global__ __launch_bounds__(128) void proj_v_kernel(const float* __restrict__ value,
                                                     const float* __restrict__ W2,
                                                     const float* __restrict__ b2,
                                                     bf16_t* __restrict__ vT) {
  const int lane = threadIdx.x & 31;
  const int wave = threadIdx.x >> 5;
  const int tilesE = E_ / 16;               // 32
  const int tilesS = S_ / 16;               // 256
  const int perBatch = tilesE * tilesS;
  int tile = blockIdx.x * 4 + wave;
  const int b  = tile / perBatch;
  const int rem = tile % perBatch;
  const int eT = rem % tilesE;
  const int sT = rem / tilesE;

  const float* aRow  = value + ((size_t)b * S_ + sT * 16 + (lane & 15)) * E_;
  const float* btRow = W2 + (size_t)(eT * 16 + (lane & 15)) * E_;

  v8f c = {};
#pragma unroll
  for (int kk = 0; kk < E_; kk += 32) {
    v16bf a  = a_frag_from_f32(aRow, kk, lane);
    v16bf bb = b_frag_from_f32(btRow, kk, lane);
    c = wmma_bf16(a, bb, c);
  }

  const int n = lane & 15;
  const float bias = b2[eT * 16 + n];
  bf16_t* outP = vT + ((size_t)b * E_ + eT * 16 + n) * S_ + sT * 16 + ((lane & 16) ? 8 : 0);
  v8bf o;
#pragma unroll
  for (int r = 0; r < 8; ++r) o[r] = (bf16_t)(c[r] + bias);
  *(v8bf*)outP = o;
}

// ---------------- phase 2: fused scores -> mask/relu -> AV ----------------
// Block = 16 waves (512 thr) covering a 64-row q tile (4 q-subtiles x 4 e-slices).
// Per 128-k superblock: wave (qt,ks) computes scores for its q-subtile over k-chunk
// ks*32, writes bf16 alpha into double-buffered LDS, barrier, then every wave
// consumes all 128 k's against its own 128-wide e slice. 64 q rows per block cuts
// per-batch L2 re-reads of k_bf / vT to ~2 GB total, below the bf16 WMMA compute
// floor, so the kernel is matrix-throughput bound.
__global__ __launch_bounds__(512) void attn_kernel(const float* __restrict__ query,
                                                   const unsigned char* __restrict__ mask,
                                                   const bf16_t* __restrict__ kbf,
                                                   const bf16_t* __restrict__ vT,
                                                   float* __restrict__ out) {
  __shared__ bf16_t alphaS[2][4][16][128];   // [buf][q-subtile][q][k]  (32 KB)

  const int lane = threadIdx.x & 31;
  const int w    = threadIdx.x >> 5;          // 0..15
  const int qt   = w >> 2;                    // q-subtile 0..3
  const int ks   = w & 3;                     // k-chunk / e-slice selector 0..3
  const int b    = blockIdx.x / (S_ / 64);
  const int qT   = blockIdx.x % (S_ / 64);
  const int qBase = qT * 64 + qt * 16;
  const int hi = (lane & 16) ? 1 : 0;
  const int n  = lane & 15;

  // Preload this wave's q-subtile A-fragments (16 frags, K = 512)
  const float* qRow = query + ((size_t)b * S_ + qBase + n) * E_;
  v16bf qf[16];
#pragma unroll
  for (int kk = 0; kk < 16; ++kk) qf[kk] = a_frag_from_f32(qRow, kk * 32, lane);

  v8f acc[8];
#pragma unroll
  for (int j = 0; j < 8; ++j) acc[j] = (v8f){};

  const int eW = ks * 128;
  const size_t maskRowBase = ((size_t)b * S_ + qBase) * (size_t)S_;
  const bf16_t* kBatch = kbf + (size_t)b * S_ * E_;
  const bf16_t* vBatch = vT + (size_t)b * E_ * S_;

  for (int kb = 0; kb < S_ / 128; ++kb) {
    const int buf = kb & 1;
    const int kcolBase = kb * 128 + ks * 32;

    // prefetch next superblock's k / v rows for this wave's chunk (global_prefetch_b8)
    if (kb + 1 < S_ / 128) {
      __builtin_prefetch(kBatch + (size_t)(kcolBase + 128 + n) * E_, 0, 1);
      __builtin_prefetch(vBatch + (size_t)(eW + n) * S_ + (kb + 1) * 128, 0, 1);
    }

    // --- scores for this wave's 32-k chunk: two 16x16 tiles ---
#pragma unroll
    for (int j = 0; j < 2; ++j) {
      v8f sc = {};
      const bf16_t* kRow = kBatch + (size_t)(kcolBase + j * 16 + n) * E_;
#pragma unroll
      for (int kk = 0; kk < 16; ++kk) {
        v16bf bb = b_frag_from_bf16(kRow, kk * 32, lane);
        sc = wmma_bf16(qf[kk], bb, sc);
      }
      // mask (with forced-true column 0) + relu -> bf16 alpha into LDS
      const int kcol = kcolBase + j * 16 + n;
#pragma unroll
      for (int r = 0; r < 8; ++r) {
        const int q = r + 8 * hi;
        const bool keep = (kcol == 0) || (mask[maskRowBase + (size_t)q * S_ + kcol] != 0);
        const float av = keep ? fmaxf(sc[r], 0.0f) : 0.0f;
        alphaS[buf][qt][q][ks * 32 + j * 16 + n] = (bf16_t)av;
      }
    }

    __syncthreads();

    // --- out += alpha(16x128) @ v(128 x eSlice) ---
#pragma unroll
    for (int kc = 0; kc < 4; ++kc) {
      const int b8 = hi ? 16 : 0;
      const v8bf lo = *(const v8bf*)(&alphaS[buf][qt][n][kc * 32 + b8]);
      const v8bf hl = *(const v8bf*)(&alphaS[buf][qt][n][kc * 32 + b8 + 8]);
      v16bf a;
#pragma unroll
      for (int i = 0; i < 8; ++i) { a[i] = lo[i]; a[8 + i] = hl[i]; }
#pragma unroll
      for (int jj = 0; jj < 8; ++jj) {
        const bf16_t* vRow = vBatch + (size_t)(eW + jj * 16 + n) * S_;
        v16bf bb = b_frag_from_bf16(vRow, kb * 128 + kc * 32, lane);
        acc[jj] = wmma_bf16(a, bb, acc[jj]);
      }
    }
  }

  // store f32 output (C layout: lane col n = e, vgpr r -> q row r + 8*hi)
#pragma unroll
  for (int jj = 0; jj < 8; ++jj)
#pragma unroll
    for (int r = 0; r < 8; ++r)
      out[((size_t)b * S_ + qBase + r + 8 * hi) * E_ + eW + jj * 16 + n] = acc[jj][r];
}

// ---------------- launch ----------------
extern "C" void kernel_launch(void* const* d_in, const int* in_sizes, int n_in,
                              void* d_out, int out_size, void* d_ws, size_t ws_size,
                              hipStream_t stream) {
  const float* query = (const float*)d_in[0];
  const float* key   = (const float*)d_in[1];
  const float* value = (const float*)d_in[2];
  const unsigned char* mask = (const unsigned char*)d_in[3];
  const float* W1 = (const float*)d_in[4];
  const float* b1 = (const float*)d_in[5];
  const float* W2 = (const float*)d_in[6];
  const float* b2 = (const float*)d_in[7];
  float* out = (float*)d_out;

  // workspace: k_bf [B,S,E] bf16 (16MB) then vT_bf [B,E,S] bf16 (16MB)
  bf16_t* kbf = (bf16_t*)d_ws;
  bf16_t* vT  = (bf16_t*)((char*)d_ws + (size_t)B_ * S_ * E_ * sizeof(bf16_t));

  const int perBatch = (E_ / 16) * (S_ / 16);     // 8192 tiles
  proj_k_kernel<<<(B_ * perBatch) / 4, dim3(128), 0, stream>>>(key, W1, b1, kbf);
  proj_v_kernel<<<(B_ * perBatch) / 4, dim3(128), 0, stream>>>(value, W2, b2, vT);
  attn_kernel<<<B_ * (S_ / 64), dim3(512), 0, stream>>>(query, mask, kbf, vT, out);
}
